// RiemannianBatchNorm_3444563771615
// MI455X (gfx1250) — matrix operand order repaired and checked by reading, not compiled, not run on hardware
//
#include <hip/hip_runtime.h>
#include <hip/hip_bf16.h>
#include <math.h>

// ---------------------------------------------------------------------------
// RiemannianBatchNorm on SPD(64) for MI455X (gfx1250, wave32, WMMA).
// Matmul-only matrix functions (Newton-Schulz sqrt, inverse scaling-and-
// squaring logm, scaling-and-squaring expm) on v_wmma_f32_16x16x32_f16.
// - all working matrices are packed f16 in LDS; every B-operand is symmetric
//   so both WMMA operands are contiguous 128-bit row reads
// - C is written back TRANSPOSED as packed v8h (1x ds_store_b128 per tile
//   half); affine epilogue amul*acc + dadd*I fused on the f32 accumulator
// - single barrier per pass (mm64t barriers at entry only)
// - x_b staged via async global->LDS; mis matrices published as f16 in the
//   workspace and async-loaded directly into the operand buffer
// ---------------------------------------------------------------------------

#define NDIM 64
#define NN   4096
#define NT   256            // 8 wave32

typedef __attribute__((ext_vector_type(16))) _Float16 v16h;
typedef __attribute__((ext_vector_type(8)))  _Float16 v8h;
typedef __attribute__((ext_vector_type(8)))  float    v8f;

// workspace layout (float units)
#define WS_MEAN  0
#define WS_TAN   NN
#define WS_VAR   (2*NN)
#define WS_MS0H  (2*NN + 16)            // f16[NN] = NN/2 float slots
#define WS_MIS0H (WS_MS0H  + NN/2)
#define WS_MIS1H (WS_MIS0H + NN/2)
#define WS_TOTAL (WS_MIS1H + NN/2 + 16)

#define HB          (NN * sizeof(_Float16))   // 8KB f16 matrix
#define FB          (NN * sizeof(float))      // 16KB f32 landing
#define SMEM_PREP   (6 * HB)                  // 48KB
#define SMEM_TAN    (FB + 7 * HB)             // 72KB -> 4 WGs/WGP
#define SMEM_UPD    (6 * HB)                  // 48KB
#define SMEM_POW    (FB + 3 * HB)             // 40KB

// ---------------------------------------------------------------------------
// reductions (leading barrier; internal barriers in the tree)
// ---------------------------------------------------------------------------
__device__ __forceinline__ float block_reduce_sum(float v, float* red) {
  red[threadIdx.x] = v;
  __syncthreads();
  #pragma unroll
  for (int off = NT / 2; off >= 1; off >>= 1) {
    if ((int)threadIdx.x < off) red[threadIdx.x] += red[threadIdx.x + off];
    __syncthreads();
  }
  float r = red[0];
  __syncthreads();
  return r;
}

__device__ __forceinline__ float block_trace_h(const _Float16* A, float* red) {
  __syncthreads();                       // A producer's stores visible
  float s = 0.f;
  if (threadIdx.x < NDIM) s = (float)A[threadIdx.x * NDIM + threadIdx.x];
  return block_reduce_sum(s, red);
}

// ---------------------------------------------------------------------------
// async global->LDS staging (ASYNCcnt path)
// ---------------------------------------------------------------------------
__device__ __forceinline__ void async_issue_f32(const float* __restrict__ gsrc,
                                                float* lds_dst) {
  const int tid = threadIdx.x;
  #pragma unroll
  for (int i = 0; i < 4; ++i) {
    unsigned long long ga =
        (unsigned long long)(const void*)(gsrc + i * 1024 + tid * 4);
    unsigned lp = (unsigned)(uintptr_t)(lds_dst + i * 1024 + tid * 4);
    asm volatile("global_load_async_to_lds_b128 %0, %1, off"
                 :: "v"(lp), "v"(ga) : "memory");
  }
}

__device__ __forceinline__ void async_issue_f16(const _Float16* __restrict__ gsrc,
                                                _Float16* lds_dst) {
  const int tid = threadIdx.x;
  #pragma unroll
  for (int i = 0; i < 2; ++i) {
    unsigned long long ga =
        (unsigned long long)(const void*)(gsrc + i * 2048 + tid * 8);
    unsigned lp = (unsigned)(uintptr_t)(lds_dst + i * 2048 + tid * 8);
    asm volatile("global_load_async_to_lds_b128 %0, %1, off"
                 :: "v"(lp), "v"(ga) : "memory");
  }
}

__device__ __forceinline__ void async_wait_all() {
  asm volatile("s_wait_asynccnt 0x0" ::: "memory");
  __syncthreads();
}

// ---------------------------------------------------------------------------
// fragment loads (packed f16 row-major LDS; B symmetric -> row reads)
// ---------------------------------------------------------------------------
__device__ __forceinline__ v16h load_frag_a(const _Float16* M, int m, int kb) {
  v8h lo = *(const v8h*)(M + m * NDIM + kb);
  v8h hi = *(const v8h*)(M + m * NDIM + kb + 16);
  return __builtin_shufflevector(lo, hi, 0,1,2,3,4,5,6,7,8,9,10,11,12,13,14,15);
}

__device__ __forceinline__ v16h load_frag_b(const _Float16* M, int n, int kb) {
  v8h lo = *(const v8h*)(M + n * NDIM + kb);
  v8h hi = *(const v8h*)(M + n * NDIM + kb + 16);
  return __builtin_shufflevector(lo, hi, 0,1,2,3,4,5,6,7,8,9,10,11,12,13,14,15);
}

// Ct = transpose(amul * (A*Bs) + dadd*I).  Bs symmetric; Ct !alias A,Bs.
// ENTRY barrier only: callers rely on the next pass's barrier to consume Ct.
__device__ __forceinline__ void mm64t(const _Float16* A, const _Float16* Bs,
                                      _Float16* Ct, float amul, float dadd) {
  __syncthreads();                         // all prior operand writes visible
  const int lane = threadIdx.x & 31;
  const int wave = threadIdx.x >> 5;
  const int tr  = wave >> 1;
  const int tc0 = (wave & 1) * 2;
  const int m   = tr  * 16 + (lane & 15);
  const int n0  = tc0 * 16 + (lane & 15);
  const int hi  = (lane >> 4) & 1;
  v8f acc0 = {}, acc1 = {};
  #pragma unroll
  for (int kt = 0; kt < 2; ++kt) {
    const int kba = kt * 32 + hi * 8;
    const int kbb = kt * 32 + hi * 8;
    v16h a  = load_frag_a(A, m, kba);
    v16h b0 = load_frag_b(Bs, n0,      kbb);
    v16h b1 = load_frag_b(Bs, n0 + 16, kbb);
    acc0 = __builtin_amdgcn_wmma_f32_16x16x32_f16(false, a, false, b0,
                                                  (short)0, acc0, false, false);
    acc1 = __builtin_amdgcn_wmma_f32_16x16x32_f16(false, a, false, b1,
                                                  (short)0, acc1, false, false);
  }
  const int mb = tr * 16 + hi * 8;
  v8h p0, p1;
  #pragma unroll
  for (int r = 0; r < 8; ++r) {
    const int mrow = mb + r;
    p0[r] = (_Float16)(acc0[r] * amul + ((mrow == n0)      ? dadd : 0.0f));
    p1[r] = (_Float16)(acc1[r] * amul + ((mrow == n0 + 16) ? dadd : 0.0f));
  }
  *(v8h*)(Ct + (n0)      * NDIM + mb) = p0;
  *(v8h*)(Ct + (n0 + 16) * NDIM + mb) = p1;
}

__device__ __forceinline__ bool is_diag(int e) { return (e >> 6) == (e & 63); }

// ---------------------------------------------------------------------------
// Newton-Schulz: AR <- sqrt(AR) in place, Zout <- invsqrt(AR).
// s0..s3 = Y/Z ping-pong; W aliases Zout during iterations.
// ---------------------------------------------------------------------------
__device__ __forceinline__ void ns_sqrt_h(_Float16* AR, _Float16* Zout,
                                          _Float16* s0, _Float16* s1,
                                          _Float16* s2, _Float16* s3,
                                          float* red, int iters) {
  const float c  = block_trace_h(AR, red) * (1.0f / NDIM);  // leading barrier
  const float rc = 1.0f / c;
  _Float16 *Y = s0, *Yn = s1, *Z = s2, *Zn = s3, *W = Zout;
  for (int e = threadIdx.x; e < NN; e += NT) {     // init pass (post-reduce
    Y[e] = (_Float16)((float)AR[e] * rc);          //  barrier already done)
    Z[e] = (_Float16)(is_diag(e) ? 1.0f : 0.0f);
  }
  for (int it = 0; it < iters; ++it) {
    mm64t(Z, Y, W, -0.5f, 1.5f);           // W  = (3I - ZY)/2   (symmetric)
    mm64t(Y, W, Yn, 1.0f, 0.0f);           // Y' = YW            (symmetric)
    mm64t(W, Z, Zn, 1.0f, 0.0f);           // Z' = WZ            (symmetric)
    _Float16* t;
    t = Y; Y = Yn; Yn = t;
    t = Z; Z = Zn; Zn = t;
  }
  __syncthreads();                         // consume last mm outputs
  const float sc = sqrtf(c), rsc = 1.0f / sc;
  for (int e = threadIdx.x; e < NN; e += NT) {
    AR[e]   = (_Float16)((float)Y[e] * sc);
    Zout[e] = (_Float16)((float)Z[e] * rsc);
  }
}

// L = logm(R); R destroyed. 3 NS square roots + order-6 Mercator series.
__device__ __forceinline__ void logm_h(_Float16* R, _Float16* L, _Float16* a,
                                       _Float16* b, _Float16* c, _Float16* d,
                                       float* red) {
  const float c0  = block_trace_h(R, red) * (1.0f / NDIM);
  const float rc0 = 1.0f / c0;
  for (int e = threadIdx.x; e < NN; e += NT)       // same-thread chain
    R[e] = (_Float16)((float)R[e] * rc0);
  for (int i = 0; i < 3; ++i)
    ns_sqrt_h(R, d, a, b, c, L, red, 8);   // R <- sqrt(R) in place
  // E = R - I in place; seed series accumulator L and power P (same-thread)
  for (int e = threadIdx.x; e < NN; e += NT) {
    const float v = (float)R[e] - (is_diag(e) ? 1.0f : 0.0f);
    R[e] = (_Float16)v;  L[e] = (_Float16)v;  a[e] = (_Float16)v;
  }
  _Float16* P = a; _Float16* Q = b;
  #pragma unroll
  for (int k = 2; k <= 6; ++k) {
    mm64t(P, R, Q, 1.0f, 0.0f);            // Q = P*E (symmetric)
    __syncthreads();                       // consume Q
    const float coef = ((k & 1) ? 1.0f : -1.0f) / (float)k;
    for (int e = threadIdx.x; e < NN; e += NT)
      L[e] = (_Float16)((float)L[e] + coef * (float)Q[e]);
    _Float16* t = P; P = Q; Q = t;
  }
  const float lc = logf(c0);
  for (int e = threadIdx.x; e < NN; e += NT)       // same-thread chain on L
    L[e] = (_Float16)(8.0f * (float)L[e] + (is_diag(e) ? lc : 0.0f));
}

// expm(N): scaling 1/16, Horner(8), 4 squarings; returns final buffer.
__device__ __forceinline__ _Float16* expm_h(_Float16* Nn, _Float16* T0,
                                            _Float16* T1) {
  __syncthreads();                         // T0/T1 reuse (WAR) + Nn producer
  for (int e = threadIdx.x; e < NN; e += NT) {
    const float nv = (float)Nn[e] * (1.0f / 16.0f);
    Nn[e] = (_Float16)nv;
    T0[e] = (_Float16)((is_diag(e) ? 1.0f : 0.0f) + nv * (1.0f / 8.0f));
  }
  _Float16* cur = T0; _Float16* nxt = T1;
  for (int k = 7; k >= 1; --k) {           // T' = I + (N*T)/k
    mm64t(Nn, cur, nxt, 1.0f / (float)k, 1.0f);
    _Float16* t = cur; cur = nxt; nxt = t;
  }
  for (int i = 0; i < 4; ++i) {            // T' = T*T
    mm64t(cur, cur, nxt, 1.0f, 0.0f);
    _Float16* t = cur; cur = nxt; nxt = t;
  }
  return cur;
}

// ---------------------------------------------------------------------------
// Phase 1: batch arithmetic-mean accumulation
// ---------------------------------------------------------------------------
__global__ __launch_bounds__(NT) void k_mean(const float* __restrict__ x,
                                             float* __restrict__ mean_acc, int Bn) {
  for (int e = threadIdx.x; e < NN; e += NT) {
    float s = 0.0f;
    for (int b = blockIdx.x; b < Bn; b += gridDim.x)
      s += x[(size_t)b * NN + e];
    atomicAdd(&mean_acc[e], s);
  }
}

// ---------------------------------------------------------------------------
// Phase 2: publish ms0 = mean^{1/2}, mis0 = mean^{-1/2} as f16 in ws
// ---------------------------------------------------------------------------
__global__ __launch_bounds__(NT) void k_prep(float* ws, int Bn) {
  extern __shared__ char smc[];
  _Float16* H = (_Float16*)smc;
  _Float16 *A = H, *B1 = H + NN, *B2 = H + 2*NN, *B3 = H + 3*NN,
           *B4 = H + 4*NN, *B5 = H + 5*NN;
  __shared__ float red[NT];
  _Float16* ms0h  = (_Float16*)(ws + WS_MS0H);
  _Float16* mis0h = (_Float16*)(ws + WS_MIS0H);
  const float rb = 1.0f / (float)Bn;
  for (int e = threadIdx.x; e < NN; e += NT)
    A[e] = (_Float16)(ws[WS_MEAN + e] * rb);
  ns_sqrt_h(A, B5, B1, B2, B3, B4, red, 12);   // A=sqrt, B5=invsqrt
  for (int e = threadIdx.x; e < NN; e += NT) { // same-thread chain
    ms0h[e]  = A[e];
    mis0h[e] = B5[e];
  }
}

// ---------------------------------------------------------------------------
// Phase 3: tangent += logm(mis0 * x_b * mis0) / B
// ---------------------------------------------------------------------------
__global__ __launch_bounds__(NT) void k_tangent(const float* __restrict__ x,
                                                float* ws, int Bn) {
  extern __shared__ char smc[];
  float* XF = (float*)smc;
  _Float16* H = (_Float16*)(smc + FB);
  _Float16 *Xh = H, *Mh = H + NN, *T1 = H + 2*NN, *R = H + 3*NN,
           *S0 = H + 4*NN, *S1 = H + 5*NN, *S2 = H + 6*NN;
  __shared__ float red[NT];
  const size_t b = blockIdx.x;
  async_issue_f32(x + b * NN, XF);
  async_issue_f16((const _Float16*)(ws + WS_MIS0H), Mh);  // direct operand
  async_wait_all();
  for (int e = threadIdx.x; e < NN; e += NT)
    Xh[e] = (_Float16)XF[e];
  mm64t(Xh, Mh, T1, 1.0f, 0.0f);   // stores (X*M)^T = M*X = T1
  mm64t(T1, Mh, R,  1.0f, 0.0f);   // stores (T1*M)^T = M*X*M = R (sym)
  logm_h(R, /*L=*/T1, S0, S1, S2, Xh, red);
  const float rb = 1.0f / (float)Bn;
  for (int e = threadIdx.x; e < NN; e += NT)   // same-thread chain on L
    atomicAdd(&ws[WS_TAN + e], (float)T1[e] * rb);
}

// ---------------------------------------------------------------------------
// Phase 4: mean1 = ms0 * expm(tangent) * ms0 ; publish mis1 = mean1^{-1/2}
// ---------------------------------------------------------------------------
__global__ __launch_bounds__(NT) void k_update(float* ws) {
  extern __shared__ char smc[];
  _Float16* H = (_Float16*)smc;
  _Float16 *A0 = H, *A1 = H + NN, *A2 = H + 2*NN, *A3 = H + 3*NN,
           *A4 = H + 4*NN, *A5 = H + 5*NN;
  __shared__ float red[NT];
  _Float16* ms0h  = (_Float16*)(ws + WS_MS0H);
  _Float16* mis1h = (_Float16*)(ws + WS_MIS1H);
  for (int e = threadIdx.x; e < NN; e += NT) A0[e] = (_Float16)ws[WS_TAN + e];
  _Float16* E = expm_h(A0, A1, A2);            // E = expm(tangent) (sym)
  _Float16* other = (E == A1) ? A2 : A1;
  for (int e = threadIdx.x; e < NN; e += NT) A3[e] = ms0h[e];
  mm64t(E, A3, other, 1.0f, 0.0f);             // other = (E*M)^T = M*E
  mm64t(other, A3, A4, 1.0f, 0.0f);            // A4 = M*E*M = mean1 (sym)
  ns_sqrt_h(A4, A5, A0, A1, A2, A3, red, 12);  // A5 = mean1^{-1/2}
  for (int e = threadIdx.x; e < NN; e += NT) mis1h[e] = A5[e];
}

// ---------------------------------------------------------------------------
// Phase 5: L_b = logm(mis1 * x_b * mis1) staged into d_out; accumulate var
// ---------------------------------------------------------------------------
__global__ __launch_bounds__(NT) void k_logxt(const float* __restrict__ x,
                                              float* ws, float* __restrict__ outL,
                                              int Bn) {
  extern __shared__ char smc[];
  float* XF = (float*)smc;
  _Float16* H = (_Float16*)(smc + FB);
  _Float16 *Xh = H, *Mh = H + NN, *T1 = H + 2*NN, *R = H + 3*NN,
           *S0 = H + 4*NN, *S1 = H + 5*NN, *S2 = H + 6*NN;
  __shared__ float red[NT];
  const size_t b = blockIdx.x;
  async_issue_f32(x + b * NN, XF);
  async_issue_f16((const _Float16*)(ws + WS_MIS1H), Mh);
  async_wait_all();
  for (int e = threadIdx.x; e < NN; e += NT)
    Xh[e] = (_Float16)XF[e];
  mm64t(Xh, Mh, T1, 1.0f, 0.0f);
  mm64t(T1, Mh, R,  1.0f, 0.0f);
  logm_h(R, /*L=*/T1, S0, S1, S2, Xh, red);
  float fro = 0.0f;
  for (int e = threadIdx.x; e < NN; e += NT) { // same-thread chain on L
    const float v = (float)T1[e];
    outL[b * NN + e] = v;
    fro += v * v;
  }
  const float tot = block_reduce_sum(fro, red);
  if (threadIdx.x == 0) atomicAdd(&ws[WS_VAR], tot / (float)Bn);
}

// ---------------------------------------------------------------------------
// Phase 6: out_b = expm(p * L_b), p = scale / (sqrt(var) + eps)
// ---------------------------------------------------------------------------
__global__ __launch_bounds__(NT) void k_pow(float* __restrict__ out,
                                            const float* ws,
                                            const float* __restrict__ scale) {
  extern __shared__ char smc[];
  float* XF = (float*)smc;
  _Float16* H = (_Float16*)(smc + FB);
  _Float16 *Nn = H, *T0 = H + NN, *T1 = H + 2*NN;
  const size_t b = blockIdx.x;
  async_issue_f32(out + b * NN, XF);
  const float p = scale[0] / (sqrtf(ws[WS_VAR]) + 1e-5f);
  async_wait_all();
  for (int e = threadIdx.x; e < NN; e += NT)
    Nn[e] = (_Float16)(XF[e] * p);
  _Float16* Tf = expm_h(Nn, T0, T1);
  __syncthreads();                             // consume final mm output
  for (int e = threadIdx.x; e < NN; e += NT) out[b * NN + e] = (float)Tf[e];
}

// ---------------------------------------------------------------------------
extern "C" void kernel_launch(void* const* d_in, const int* in_sizes, int n_in,
                              void* d_out, int out_size, void* d_ws, size_t ws_size,
                              hipStream_t stream) {
  const float* x     = (const float*)d_in[0];
  const float* scale = (const float*)d_in[1];
  float* out = (float*)d_out;
  float* ws  = (float*)d_ws;
  const int Bn = in_sizes[0] / NN;
  if (Bn <= 0) return;

  hipMemsetAsync(d_ws, 0, (size_t)WS_TOTAL * sizeof(float), stream);

  k_mean   <<<64, NT, 0,         stream>>>(x, ws + WS_MEAN, Bn);
  k_prep   <<<1,  NT, SMEM_PREP, stream>>>(ws, Bn);
  k_tangent<<<Bn, NT, SMEM_TAN,  stream>>>(x, ws, Bn);
  k_update <<<1,  NT, SMEM_UPD,  stream>>>(ws);
  k_logxt  <<<Bn, NT, SMEM_TAN,  stream>>>(x, ws, out, Bn);
  k_pow    <<<Bn, NT, SMEM_POW,  stream>>>(out, ws, scale);
}
